// CondLinSinkhornPRModel_78520592105610
// MI455X (gfx1250) — compile-verified, matrix operand order checked by reading, not run on hardware
//
#include <hip/hip_runtime.h>

// ---------------- problem constants (from reference) ----------------
#define D_   128
#define B_   16
#define ND_  2048
#define NS_  512
#define EPS_     0.0025f      // BLUR^P = 0.05^2
#define INV_EPS_ 400.0f
#define NITERS_  20
#define SCALE_   10.0f

typedef __attribute__((ext_vector_type(16))) __bf16 bf16x16;
typedef __attribute__((ext_vector_type(8)))  float  floatx8;

// float -> bf16 (RNE) stored as ushort
__device__ __forceinline__ unsigned short f2bf(float x) {
    unsigned u = __float_as_uint(x);
    unsigned r = u + 0x7FFFu + ((u >> 16) & 1u);
    return (unsigned short)(r >> 16);
}

// ---------------- small setup kernels ----------------

// dbar[b,k] = mean_n d[b,n,k]
__global__ void mean_kernel(const float* __restrict__ d, float* __restrict__ dbar) {
    int b = blockIdx.x, k = threadIdx.x;
    const float* p = d + (size_t)b * ND_ * D_ + k;
    float s = 0.f;
    for (int n = 0; n < ND_; ++n) s += p[(size_t)n * D_];
    dbar[b * D_ + k] = s * (1.0f / ND_);
}

// M[b,o] = relu(dbar[b,:] @ W[:,o] + bias[o]),  o in [0, D*D)
__global__ void mrelu_kernel(const float* __restrict__ dbar, const float* __restrict__ W,
                             const float* __restrict__ bias, float* __restrict__ M) {
    int b = blockIdx.y;
    int o = blockIdx.x * blockDim.x + threadIdx.x;
    __shared__ float db[D_];
    if (threadIdx.x < D_) db[threadIdx.x] = dbar[b * D_ + threadIdx.x];
    __syncthreads();
    float acc = bias[o];
    for (int k = 0; k < D_; ++k) acc += db[k] * W[(size_t)k * (D_ * D_) + o];
    M[(size_t)b * D_ * D_ + o] = fmaxf(acc, 0.f);
}

// t[b,row,:] = X[b,row,:] @ M[b];  store bf16 + squared norm
__global__ void transform_kernel(const float* __restrict__ X, const float* __restrict__ M,
                                 unsigned short* __restrict__ Th, float* __restrict__ sq,
                                 int Nrows) {
    int b = blockIdx.y, row = blockIdx.x, e = threadIdx.x;
    const float* xr = X + ((size_t)b * Nrows + row) * D_;
    const float* Mb = M + (size_t)b * D_ * D_;
    __shared__ float xs[D_];
    __shared__ float red[D_];
    xs[e] = xr[e];
    __syncthreads();
    float acc = 0.f;
    for (int k = 0; k < D_; ++k) acc += xs[k] * Mb[k * D_ + e];
    Th[((size_t)b * Nrows + row) * D_ + e] = f2bf(acc);
    red[e] = acc * acc;
    __syncthreads();
    for (int s = D_ / 2; s > 0; s >>= 1) { if (e < s) red[e] += red[e + s]; __syncthreads(); }
    if (e == 0) sq[(size_t)b * Nrows + row] = red[0];
}

__global__ void log_kernel(const float* __restrict__ in, float* __restrict__ out, int n) {
    int i = blockIdx.x * blockDim.x + threadIdx.x;
    if (i < n) out[i] = __logf(in[i]);
}

__global__ void zero_kernel(float* __restrict__ p, int n) {
    int i = blockIdx.x * blockDim.x + threadIdx.x;
    if (i < n) p[i] = 0.f;
}

// ---------------- fused WMMA Gram + online logsumexp pass ----------------
// out[row] = 0.5*|x_row|^2 - EPS * lse_j( lw[j] + (pot[j] - 0.5*|y_j|^2)/EPS + (x_row . y_j)/EPS )
// if mixOld: out[row] = 0.5*(mixOld[row] + out[row])   (symmetric fixed-point step)
// 1 wave per 16-row X tile, 4 waves / block, column term staged in LDS,
// B fragments double-buffered so global loads overlap WMMA + exp VALU work.
// Requires Nx,Ny multiples of 16, Ny <= 2048.
__global__ __launch_bounds__(128) void lse_pass_kernel(
    const unsigned short* __restrict__ Xh, const float* __restrict__ xsq, int Nx,
    const unsigned short* __restrict__ Yh, const float* __restrict__ ysq, int Ny,
    const float* __restrict__ lw, const float* __restrict__ pot,
    const float* __restrict__ mixOld, float* __restrict__ outPot) {
    const int b     = blockIdx.y;
    const int wave  = threadIdx.x >> 5;
    const int lane  = threadIdx.x & 31;
    const int half  = lane >> 4;          // 0: K 0..15 / rows 0..7, 1: K 16..31 / rows 8..15
    const int l16   = lane & 15;
    const int rowbase = (blockIdx.x * 4 + wave) * 16;

    const unsigned short* Xb = Xh + (size_t)b * Nx * D_;
    const unsigned short* Yb = Yh + (size_t)b * Ny * D_;
    const float* xsqb = xsq + (size_t)b * Nx;
    const float* ysqb = ysq + (size_t)b * Ny;
    const float* lwb  = lw  + (size_t)b * Ny;
    const float* potb = pot + (size_t)b * Ny;

    // column term, loop-invariant per launch: shared by all 4 waves
    __shared__ float cterm[2048];
    for (int idx = threadIdx.x; idx < Ny; idx += 128)
        cterm[idx] = lwb[idx] + (potb[idx] - 0.5f * ysqb[idx]) * INV_EPS_;

    // A fragments: 4 chunks of K=32, contiguous 32B per lane
    bf16x16 afr[4];
    const unsigned short* xrow = Xb + (size_t)(rowbase + l16) * D_ + half * 16;
#pragma unroll
    for (int kb = 0; kb < 4; ++kb) afr[kb] = *(const bf16x16*)(xrow + kb * 32);

    float rmax[8], rsum[8];
#pragma unroll
    for (int r = 0; r < 8; ++r) { rmax[r] = -3.0e38f; rsum[r] = 0.f; }

    __syncthreads();   // cterm ready

    // B fragment double buffer: prefetch tile jb=0
    const unsigned short* ybase = Yb + (size_t)l16 * D_ + half * 16;
    bf16x16 bnx[4];
#pragma unroll
    for (int kb = 0; kb < 4; ++kb) bnx[kb] = *(const bf16x16*)(ybase + kb * 32);

    for (int jb = 0; jb < Ny; jb += 16) {
        bf16x16 bcur[4];
#pragma unroll
        for (int kb = 0; kb < 4; ++kb) bcur[kb] = bnx[kb];
        if (jb + 16 < Ny) {   // issue next tile's loads before consuming current
            const unsigned short* ynext = ybase + (size_t)(jb + 16) * D_;
#pragma unroll
            for (int kb = 0; kb < 4; ++kb) bnx[kb] = *(const bf16x16*)(ynext + kb * 32);
        }
        floatx8 c = {0.f, 0.f, 0.f, 0.f, 0.f, 0.f, 0.f, 0.f};
#pragma unroll
        for (int kb = 0; kb < 4; ++kb)
            c = __builtin_amdgcn_wmma_f32_16x16x32_bf16(false, afr[kb], false, bcur[kb],
                                                        (short)0, c, false, false);
        const float ct = cterm[jb + l16];
#pragma unroll
        for (int r = 0; r < 8; ++r) {
            float t  = ct + c[r] * INV_EPS_;
            float nm = fmaxf(rmax[r], t);
            rsum[r]  = rsum[r] * __expf(rmax[r] - nm) + __expf(t - nm);
            rmax[r]  = nm;
        }
    }
    // merge 16 lanes of each half (each output row lives in one 16-lane group)
#pragma unroll
    for (int r = 0; r < 8; ++r) {
        float m = rmax[r], s = rsum[r];
        for (int off = 1; off < 16; off <<= 1) {
            float om = __shfl_xor(m, off, 16);
            float os = __shfl_xor(s, off, 16);
            float nm = fmaxf(m, om);
            s = s * __expf(m - nm) + os * __expf(om - nm);
            m = nm;
        }
        rmax[r] = m; rsum[r] = s;
    }
    if (l16 == 0) {
#pragma unroll
        for (int r = 0; r < 8; ++r) {
            int row = rowbase + half * 8 + r;
            float val = 0.5f * xsqb[row] - EPS_ * (__logf(rsum[r]) + rmax[r]);
            if (mixOld) val = 0.5f * (mixOld[(size_t)b * Nx + row] + val);
            outPot[(size_t)b * Nx + row] = val;
        }
    }
}

// ---------------- final reduction ----------------
// dist2-dist1 = sum h*(f2-f1) + sum hj*(g2-qj) - sum hi*(g1-qi)   (p cancels)
__global__ void final_kernel(const float* __restrict__ h, const float* __restrict__ hi,
                             const float* __restrict__ hj,
                             const float* __restrict__ f1, const float* __restrict__ f2,
                             const float* __restrict__ g1, const float* __restrict__ g2,
                             const float* __restrict__ qi, const float* __restrict__ qj,
                             float* __restrict__ out) {
    int b = blockIdx.x, t = threadIdx.x;
    float acc = 0.f;
    for (int n = t; n < ND_; n += blockDim.x)
        acc += h[b * ND_ + n] * (f2[b * ND_ + n] - f1[b * ND_ + n]);
    for (int j = t; j < NS_; j += blockDim.x) {
        acc += hj[b * NS_ + j] * (g2[b * NS_ + j] - qj[b * NS_ + j]);
        acc -= hi[b * NS_ + j] * (g1[b * NS_ + j] - qi[b * NS_ + j]);
    }
    __shared__ float red[256];
    red[t] = acc;
    __syncthreads();
    for (int s = 128; s > 0; s >>= 1) { if (t < s) red[t] += red[t + s]; __syncthreads(); }
    if (t == 0) out[b] = 1.f / (1.f + __expf(-SCALE_ * red[0]));
}

// ---------------- host orchestration ----------------
extern "C" void kernel_launch(void* const* d_in, const int* in_sizes, int n_in,
                              void* d_out, int out_size, void* d_ws, size_t ws_size,
                              hipStream_t stream) {
    (void)in_sizes; (void)n_in; (void)out_size; (void)ws_size;
    const float* d  = (const float*)d_in[0];
    const float* si = (const float*)d_in[1];
    const float* sj = (const float*)d_in[2];
    const float* h  = (const float*)d_in[3];
    const float* hi = (const float*)d_in[4];
    const float* hj = (const float*)d_in[5];
    const float* W  = (const float*)d_in[6];
    const float* bb = (const float*)d_in[7];
    float* out = (float*)d_out;

    // bump allocator on workspace (~15 MB total)
    char* p = (char*)d_ws;
    auto alloc = [&](size_t bytes) -> void* {
        void* r = (void*)p;
        p += (bytes + 255) & ~(size_t)255;
        return r;
    };
    float*          dbar = (float*)alloc((size_t)B_ * D_ * 4);
    float*          M    = (float*)alloc((size_t)B_ * D_ * D_ * 4);
    unsigned short* tdh  = (unsigned short*)alloc((size_t)B_ * ND_ * D_ * 2);
    unsigned short* tsih = (unsigned short*)alloc((size_t)B_ * NS_ * D_ * 2);
    unsigned short* tsjh = (unsigned short*)alloc((size_t)B_ * NS_ * D_ * 2);
    float* ntd  = (float*)alloc((size_t)B_ * ND_ * 4);
    float* ntsi = (float*)alloc((size_t)B_ * NS_ * 4);
    float* ntsj = (float*)alloc((size_t)B_ * NS_ * 4);
    float* lh   = (float*)alloc((size_t)B_ * ND_ * 4);
    float* lhi  = (float*)alloc((size_t)B_ * NS_ * 4);
    float* lhj  = (float*)alloc((size_t)B_ * NS_ * 4);
    float* f1   = (float*)alloc((size_t)B_ * ND_ * 4);
    float* f2   = (float*)alloc((size_t)B_ * ND_ * 4);
    float* g1   = (float*)alloc((size_t)B_ * NS_ * 4);
    float* g2   = (float*)alloc((size_t)B_ * NS_ * 4);
    float* qiA  = (float*)alloc((size_t)B_ * NS_ * 4);
    float* qiB  = (float*)alloc((size_t)B_ * NS_ * 4);
    float* qjA  = (float*)alloc((size_t)B_ * NS_ * 4);
    float* qjB  = (float*)alloc((size_t)B_ * NS_ * 4);

    // setup
    mean_kernel<<<B_, D_, 0, stream>>>(d, dbar);
    mrelu_kernel<<<dim3((D_ * D_) / 128, B_), 128, 0, stream>>>(dbar, W, bb, M);
    transform_kernel<<<dim3(ND_, B_), D_, 0, stream>>>(d,  M, tdh,  ntd,  ND_);
    transform_kernel<<<dim3(NS_, B_), D_, 0, stream>>>(si, M, tsih, ntsi, NS_);
    transform_kernel<<<dim3(NS_, B_), D_, 0, stream>>>(sj, M, tsjh, ntsj, NS_);
    log_kernel<<<(B_ * ND_ + 255) / 256, 256, 0, stream>>>(h,  lh,  B_ * ND_);
    log_kernel<<<(B_ * NS_ + 255) / 256, 256, 0, stream>>>(hi, lhi, B_ * NS_);
    log_kernel<<<(B_ * NS_ + 255) / 256, 256, 0, stream>>>(hj, lhj, B_ * NS_);
    zero_kernel<<<(B_ * ND_ + 255) / 256, 256, 0, stream>>>(f1,  B_ * ND_);
    zero_kernel<<<(B_ * ND_ + 255) / 256, 256, 0, stream>>>(f2,  B_ * ND_);
    zero_kernel<<<(B_ * NS_ + 255) / 256, 256, 0, stream>>>(qiA, B_ * NS_);
    zero_kernel<<<(B_ * NS_ + 255) / 256, 256, 0, stream>>>(qjA, B_ * NS_);

    const dim3 gridBig(ND_ / 16 / 4, B_);   // X = td       (32, 16)
    const dim3 gridSml(NS_ / 16 / 4, B_);   // X = tsi/tsj  (8, 16)

    for (int it = 0; it < NITERS_; ++it) {
        // g1[j] = -eps lse_n( log h[n] + (f1[n]-C(td_n,tsi_j))/eps )
        lse_pass_kernel<<<gridSml, 128, 0, stream>>>(tsih, ntsi, NS_, tdh, ntd, ND_,
                                                     lh, f1, nullptr, g1);
        // f1[n] = -eps lse_j( log hi[j] + (g1[j]-C)/eps )
        lse_pass_kernel<<<gridBig, 128, 0, stream>>>(tdh, ntd, ND_, tsih, ntsi, NS_,
                                                     lhi, g1, nullptr, f1);
        lse_pass_kernel<<<gridSml, 128, 0, stream>>>(tsjh, ntsj, NS_, tdh, ntd, ND_,
                                                     lh, f2, nullptr, g2);
        lse_pass_kernel<<<gridBig, 128, 0, stream>>>(tdh, ntd, ND_, tsjh, ntsj, NS_,
                                                     lhj, g2, nullptr, f2);
        // symmetric debias potentials (double-buffered fixed point)
        lse_pass_kernel<<<gridSml, 128, 0, stream>>>(tsih, ntsi, NS_, tsih, ntsi, NS_,
                                                     lhi, qiA, qiA, qiB);
        { float* t = qiA; qiA = qiB; qiB = t; }
        lse_pass_kernel<<<gridSml, 128, 0, stream>>>(tsjh, ntsj, NS_, tsjh, ntsj, NS_,
                                                     lhj, qjA, qjA, qjB);
        { float* t = qjA; qjA = qjB; qjB = t; }
    }

    final_kernel<<<B_, 256, 0, stream>>>(h, hi, hj, f1, f2, g1, g2, qiA, qjA, out);
}
